// PointNetPPBoxEstimation_8890582303362
// MI455X (gfx1250) — matrix operand order repaired and checked by reading, not compiled, not run on hardware
//
#include <hip/hip_runtime.h>
#include <cstdint>

// PointNet++ box-estimation pipeline for gfx1250 (MI455X), wave32 + WMMA.
// All MLP layers run as v_wmma_f32_16x16x32_f16 tile GEMMs with BN folded
// into a per-channel f32 scale/shift epilogue. Grouped activations live in
// a 48KB LDS arena per workgroup; weights are pre-converted to padded f16.

typedef __attribute__((ext_vector_type(16))) _Float16 v16h;
typedef __attribute__((ext_vector_type(8)))  _Float16 v8h;
typedef __attribute__((ext_vector_type(8)))  float    v8f;

#define BN_EPS 1e-5f

// ---------------- WMMA fragment helpers (CDNA5 ISA 7.12.2, wave32) ---------

__device__ __forceinline__ v8f wmma_f16(v16h a, v16h b, v8f c) {
  // 8 args: (neg_a, A, neg_b, B, c_mod, C, reuse_a, reuse_b)
  return __builtin_amdgcn_wmma_f32_16x16x32_f16(false, a, false, b, (short)0, c,
                                                false, false);
}

// A: row-major [16 rows][lda] f16 tile starting at Arow. A layout used here (contiguous k, same as B):
// lanes 0-15 -> M=lane, elems 0..15 = K(kbase..+15)
// lanes 16-31 -> M=lane-16, elems 0..15 = K(kbase+16..+31)
__device__ __forceinline__ v16h load_a_frag(const _Float16* Arow, int lda, int kbase) {
  const int lane = threadIdx.x & 31;
  const _Float16* p = Arow + (size_t)(lane & 15) * lda + kbase + ((lane >> 4) << 4);
  v8h lo = *(const v8h*)(p);
  v8h hv = *(const v8h*)(p + 8);
  v16h r;
#pragma unroll
  for (int i = 0; i < 8; ++i) { r[i] = lo[i]; r[i + 8] = hv[i]; }
  return r;
}

// B element (k, n) = W[n][k]  (W is [O][ldk] f16, row = output channel).
// lanes 0-15 -> N=lane, K = kbase + e; lanes 16-31 -> N=lane-16, K = kbase+16+e.
__device__ __forceinline__ v16h load_b_frag(const _Float16* W, int ldk, int nbase, int kbase) {
  const int lane = threadIdx.x & 31;
  return *(const v16h*)(W + (size_t)(nbase + (lane & 15)) * ldk + kbase + ((lane >> 4) << 4));
}

// Fused linear (+BN folded scale/shift) (+ReLU): LDS f16 in -> LDS f16 out.
// rows, Cout multiples of 16; Kpad multiple of 32.
__device__ void mlp_layer(const _Float16* __restrict__ inL, int rows, int Kpad,
                          const _Float16* __restrict__ Wg,
                          const float* __restrict__ sc_, const float* __restrict__ sh_,
                          int Cout, _Float16* __restrict__ outL, int relu) {
  const int wave   = threadIdx.x >> 5;
  const int nwaves = blockDim.x >> 5;
  const int lane = threadIdx.x & 31;
  const int hi = lane >> 4;
  const int ln = lane & 15;
  const int mtn = rows >> 4, ntn = Cout >> 4;
  for (int tile = wave; tile < mtn * ntn; tile += nwaves) {
    const int mt = tile % mtn, nt = tile / mtn;
    v8f acc = {};
    for (int kt = 0; kt < Kpad; kt += 32) {
      if (kt + 32 < Kpad)  // stream next weight slice through the cache
        __builtin_prefetch(Wg + (size_t)(nt * 16 + ln) * Kpad + kt + 32, 0, 0);
      v16h a = load_a_frag(inL + (size_t)(mt * 16) * Kpad, Kpad, kt);
      v16h b = load_b_frag(Wg, Kpad, nt * 16, kt);
      acc = wmma_f16(a, b, acc);
    }
    const int col = nt * 16 + ln;
    const float sc = sc_[col], sh = sh_[col];
#pragma unroll
    for (int r = 0; r < 8; ++r) {
      const int m = mt * 16 + r + hi * 8;   // D layout: M = r + 8*hi
      float v = acc[r] * sc + sh;
      if (relu && v < 0.f) v = 0.f;
      outL[(size_t)m * Cout + col] = (_Float16)v;
    }
  }
}

// ---------------- weight/BN prep: f32 -> padded f16 + (scale, shift) -------

__global__ void prep_layer_kernel(const float* __restrict__ W, const float* __restrict__ bias,
                                  const float* __restrict__ gamma, const float* __restrict__ beta,
                                  const float* __restrict__ mean, const float* __restrict__ var,
                                  int Oreal, int Opad, int K, int Kpad,
                                  _Float16* __restrict__ Wf, float* __restrict__ s,
                                  float* __restrict__ t, int hasBN) {
  const int i = blockIdx.x * blockDim.x + threadIdx.x;
  if (i < Opad * Kpad) {
    const int o = i / Kpad, k = i % Kpad;
    Wf[i] = (o < Oreal && k < K) ? (_Float16)W[(size_t)o * K + k] : (_Float16)0.f;
  }
  if (i < Opad) {
    float sc = 0.f, sh = 0.f;
    if (i < Oreal) {
      if (hasBN) {
        sc = gamma[i] * rsqrtf(var[i] + BN_EPS);
        sh = (bias[i] - mean[i]) * sc + beta[i];
      } else { sc = 1.f; sh = bias[i]; }
    }
    s[i] = sc; t[i] = sh;
  }
}

// ---------------- layout / gather kernels ----------------------------------

__global__ void transpose_pts_kernel(const float* __restrict__ pts, float* __restrict__ xyz,
                                     int B, int N) {
  const int i = blockIdx.x * blockDim.x + threadIdx.x;
  if (i >= B * N * 3) return;
  const int c = i % 3, n = (i / 3) % N, b = i / (3 * N);
  xyz[i] = pts[(size_t)b * 3 * N + (size_t)c * N + n];
}

__global__ void gather_xyz_kernel(const float* __restrict__ xyz, const int* __restrict__ idx,
                                  float* __restrict__ out, int B, int N, int S) {
  const int i = blockIdx.x * blockDim.x + threadIdx.x;
  if (i >= B * S * 3) return;
  const int c = i % 3, s = (i / 3) % S, b = i / (3 * S);
  out[i] = xyz[((size_t)b * N + idx[b * S + s]) * 3 + c];
}

// ---------------- farthest point sampling (one WG per batch) ---------------

__global__ __launch_bounds__(256)
void fps_kernel(const float* __restrict__ xyz, int N, int npoint, int* __restrict__ idxOut) {
  __shared__ float dist[2048];
  __shared__ float rv[256];
  __shared__ int   ri[256];
  __shared__ int   farS;
  const int b = blockIdx.x;
  for (int n = threadIdx.x; n < N; n += blockDim.x) dist[n] = 1e10f;
  if (threadIdx.x == 0) farS = 0;
  __syncthreads();
  for (int it = 0; it < npoint; ++it) {
    const int far = farS;
    if (threadIdx.x == 0) idxOut[b * npoint + it] = far;  // emit BEFORE update (lax.scan)
    const float cx = xyz[((size_t)b * N + far) * 3 + 0];
    const float cy = xyz[((size_t)b * N + far) * 3 + 1];
    const float cz = xyz[((size_t)b * N + far) * 3 + 2];
    float bv = -1.f; int bi = 0;
    for (int n = threadIdx.x; n < N; n += blockDim.x) {
      const float dx = xyz[((size_t)b * N + n) * 3 + 0] - cx;
      const float dy = xyz[((size_t)b * N + n) * 3 + 1] - cy;
      const float dz = xyz[((size_t)b * N + n) * 3 + 2] - cz;
      const float d = dx * dx + dy * dy + dz * dz;
      float dd = dist[n]; dd = (d < dd) ? d : dd; dist[n] = dd;
      if (dd > bv) { bv = dd; bi = n; }
    }
    rv[threadIdx.x] = bv; ri[threadIdx.x] = bi;
    __syncthreads();
    for (int o = 128; o > 0; o >>= 1) {
      if (threadIdx.x < (unsigned)o) {
        const float ov = rv[threadIdx.x + o]; const int oi = ri[threadIdx.x + o];
        if (ov > rv[threadIdx.x] || (ov == rv[threadIdx.x] && oi < ri[threadIdx.x])) {
          rv[threadIdx.x] = ov; ri[threadIdx.x] = oi;
        }
      }
      __syncthreads();
    }
    if (threadIdx.x == 0) farS = ri[0];
    __syncthreads();
  }
}

// ---------------- fused SA stage: ball-query + group + 3x MLP + max --------
// One workgroup (128 threads = 4 waves) per (batch, centroid) group.
// LDS arena ping-pong: p0[0,10240) featIn, p1[10240,18432) act1,
//                      p2[0,8192) act2, p3[8192,24576) act3.
__global__ __launch_bounds__(128)
void sa_fused_kernel(const float* __restrict__ xyz, const float* __restrict__ pfeat, int Cpts,
                     const float* __restrict__ new_xyz, int B, int N, int S,
                     int nsample, float r2, int Cin, int KinPad,
                     const _Float16* W1, const float* s1, const float* t1, int C1,
                     const _Float16* W2, const float* s2, const float* t2, int C2,
                     const _Float16* W3, const float* s3, const float* t3, int C3,
                     float* __restrict__ out, int groupAll) {
  __shared__ _Float16 arena[24576];   // 48 KB
  __shared__ float cent[3];
  __shared__ int sel[64];
  __shared__ int pref[129];

  const int b  = blockIdx.x / S;
  const int si = blockIdx.x % S;
  const int tid = threadIdx.x;

  if (groupAll) {
    for (int j = tid; j < nsample; j += blockDim.x) sel[j] = j;
    if (tid < 3) cent[tid] = 0.f;
  } else {
    if (tid < 3) cent[tid] = new_xyz[((size_t)b * S + si) * 3 + tid];
    __syncthreads();
    int chunk = N / (int)blockDim.x; if (chunk < 1) chunk = 1;
    const int beg = tid * chunk;
    int end = beg + chunk; if (end > N) end = N;
    int cnt = 0;
    for (int n = beg; n < end; ++n) {
      const float dx = xyz[((size_t)b * N + n) * 3 + 0] - cent[0];
      const float dy = xyz[((size_t)b * N + n) * 3 + 1] - cent[1];
      const float dz = xyz[((size_t)b * N + n) * 3 + 2] - cent[2];
      if (dx * dx + dy * dy + dz * dz <= r2) ++cnt;
    }
    pref[tid] = cnt;
    __syncthreads();
    if (tid == 0) {                       // ordered compaction -> lowest indices win
      int run = 0;
      for (int i = 0; i < (int)blockDim.x; ++i) { const int c = pref[i]; pref[i] = run; run += c; }
      pref[blockDim.x] = run;
    }
    __syncthreads();
    int pos = pref[tid];
    for (int n = beg; n < end && pos < nsample; ++n) {
      const float dx = xyz[((size_t)b * N + n) * 3 + 0] - cent[0];
      const float dy = xyz[((size_t)b * N + n) * 3 + 1] - cent[1];
      const float dz = xyz[((size_t)b * N + n) * 3 + 2] - cent[2];
      if (dx * dx + dy * dy + dz * dz <= r2) { sel[pos] = n; ++pos; }
    }
    __syncthreads();
    if (tid == 0) {
      int tot = pref[blockDim.x]; if (tot > nsample) tot = nsample;
      const int f = (tot > 0) ? sel[0] : 0;
      for (int j = tot; j < nsample; ++j) sel[j] = f;  // pad with first index
    }
  }
  __syncthreads();

  _Float16* p0 = arena;
  _Float16* p1 = arena + 10240;
  _Float16* p2 = arena;
  _Float16* p3 = arena + 8192;

  // stage grouped features (centered xyz ++ point feats, zero-padded) as f16
  const int tot = nsample * KinPad;
  for (int i = tid; i < tot; i += blockDim.x) {
    const int r = i / KinPad, k = i % KinPad;
    float v = 0.f;
    const int n = sel[r];
    if (k < 3)          v = xyz[((size_t)b * N + n) * 3 + k] - cent[k];
    else if (k < Cin)   v = pfeat[((size_t)b * N + n) * Cpts + (k - 3)];
    p0[i] = (_Float16)v;
  }
  __syncthreads();

  mlp_layer(p0, nsample, KinPad, W1, s1, t1, C1, p1, 1); __syncthreads();
  mlp_layer(p1, nsample, C1,     W2, s2, t2, C2, p2, 1); __syncthreads();
  mlp_layer(p2, nsample, C2,     W3, s3, t3, C3, p3, 1); __syncthreads();

  // max over the sample axis
  for (int c = tid; c < C3; c += blockDim.x) {
    float m = -1e30f;
    for (int r = 0; r < nsample; ++r) {
      const float v = (float)p3[(size_t)r * C3 + c];
      m = (v > m) ? v : m;
    }
    out[((size_t)b * S + si) * C3 + c] = m;
  }
}

// ---------------- FC head GEMM (WMMA tiles, grid-strided waves) ------------

__global__ void build_fc_in_kernel(const float* __restrict__ gfeat, const float* __restrict__ oh,
                                   _Float16* __restrict__ out, int B) {
  const int i = blockIdx.x * blockDim.x + threadIdx.x;
  if (i >= B * 544) return;
  const int k = i % 544, b = i / 544;
  float v = 0.f;
  if (k < 512)      v = gfeat[b * 512 + k];
  else if (k < 514) v = oh[b * 2 + (k - 512)];
  out[i] = (_Float16)v;
}

__global__ __launch_bounds__(128)
void fc_gemm_kernel(const _Float16* __restrict__ A, int M, int Kpad,
                    const _Float16* __restrict__ Wf, const float* __restrict__ s,
                    const float* __restrict__ t, int Opad, int Oreal,
                    float* __restrict__ out32, _Float16* __restrict__ out16, int relu) {
  const int lane = threadIdx.x & 31;
  const int hi = lane >> 4;
  const int ln = lane & 15;
  const int mtn = M >> 4, ntn = Opad >> 4;
  const int tiles = mtn * ntn;
  const int gw = (blockIdx.x * blockDim.x + threadIdx.x) >> 5;
  const int nw = (gridDim.x * blockDim.x) >> 5;
  for (int tile = gw; tile < tiles; tile += nw) {
    const int mt = tile % mtn, nt = tile / mtn;
    v8f acc = {};
    for (int kt = 0; kt < Kpad; kt += 32) {
      v16h a = load_a_frag(A + (size_t)(mt * 16) * Kpad, Kpad, kt);
      v16h b = load_b_frag(Wf, Kpad, nt * 16, kt);
      acc = wmma_f16(a, b, acc);
    }
    const int col = nt * 16 + ln;
    const float sc = s[col], sh = t[col];
#pragma unroll
    for (int r = 0; r < 8; ++r) {
      const int m = mt * 16 + r + hi * 8;
      float v = acc[r] * sc + sh;
      if (relu && v < 0.f) v = 0.f;
      if (out16) out16[(size_t)m * Opad + col] = (_Float16)v;
      if (out32 && col < Oreal) out32[(size_t)m * Oreal + col] = v;
    }
  }
}

// ---------------- host orchestration ---------------------------------------

extern "C" void kernel_launch(void* const* d_in, const int* in_sizes, int n_in,
                              void* d_out, int out_size, void* d_ws, size_t ws_size,
                              hipStream_t stream) {
  (void)in_sizes; (void)n_in; (void)out_size; (void)ws_size;
  const int B = 128, N = 2048;
  const float* pts     = (const float*)d_in[0];
  const float* one_hot = (const float*)d_in[1];
  auto getf = [&](int i) { return (const float*)d_in[i]; };

  // Flattened input layout (recursive insertion order):
  // 0:pts 1:one_hot; sa{1,2,3}_params: 3 layers x [W,b,gamma,beta,mean,var];
  // fc: fc1_W fc1_b bn1{g,b,m,v} fc2_W fc2_b bn2{g,b,m,v} fc3_W fc3_b
  struct Ld { const float *W, *b, *g, *be, *mn, *vr; int Oreal, K; };
  Ld L[12];
  const int dims[12][2] = {
    {64, 3}, {64, 64}, {128, 64},       // SA1
    {128, 131}, {128, 128}, {256, 128}, // SA2
    {256, 259}, {256, 256}, {512, 256}, // SA3
    {512, 514}, {256, 512}, {59, 256}   // FC1..FC3
  };
  for (int g = 0; g < 3; ++g)
    for (int l = 0; l < 3; ++l) {
      const int base = 2 + g * 18 + l * 6;
      const int li = g * 3 + l;
      L[li] = { getf(base + 0), getf(base + 1), getf(base + 2), getf(base + 3),
                getf(base + 4), getf(base + 5), dims[li][0], dims[li][1] };
    }
  L[9]  = { getf(56), getf(57), getf(58), getf(59), getf(60), getf(61), 512, 514 };
  L[10] = { getf(62), getf(63), getf(64), getf(65), getf(66), getf(67), 256, 512 };
  L[11] = { getf(68), getf(69), nullptr, nullptr, nullptr, nullptr, 59, 256 };

  // workspace carve-out (256B aligned)
  char* base = (char*)d_ws;
  size_t off = 0;
  auto alloc = [&](size_t bytes) -> void* {
    off = (off + 255) & ~(size_t)255;
    void* p = base + off; off += bytes; return p;
  };
  float*    xyz   = (float*)   alloc((size_t)B * N * 3 * sizeof(float));
  int*      fps1  = (int*)     alloc((size_t)B * 128 * sizeof(int));
  float*    l1xyz = (float*)   alloc((size_t)B * 128 * 3 * sizeof(float));
  float*    l1pts = (float*)   alloc((size_t)B * 128 * 128 * sizeof(float));
  int*      fps2  = (int*)     alloc((size_t)B * 32 * sizeof(int));
  float*    l2xyz = (float*)   alloc((size_t)B * 32 * 3 * sizeof(float));
  float*    l2pts = (float*)   alloc((size_t)B * 32 * 256 * sizeof(float));
  float*    gfeat = (float*)   alloc((size_t)B * 512 * sizeof(float));
  _Float16* fcin  = (_Float16*)alloc((size_t)B * 544 * sizeof(_Float16));
  _Float16* fc1o  = (_Float16*)alloc((size_t)B * 512 * sizeof(_Float16));
  _Float16* fc2o  = (_Float16*)alloc((size_t)B * 256 * sizeof(_Float16));

  _Float16* Wf[12]; float* Sc[12]; float* Sh[12];
  int Opad[12], Kpad[12];
  for (int i = 0; i < 12; ++i) {
    Opad[i] = (L[i].Oreal + 15) & ~15;
    Kpad[i] = (L[i].K + 31) & ~31;
    Wf[i] = (_Float16*)alloc((size_t)Opad[i] * Kpad[i] * sizeof(_Float16));
    Sc[i] = (float*)alloc((size_t)Opad[i] * sizeof(float));
    Sh[i] = (float*)alloc((size_t)Opad[i] * sizeof(float));
  }

  for (int i = 0; i < 12; ++i) {
    const int tot = Opad[i] * Kpad[i];
    prep_layer_kernel<<<dim3((tot + 255) / 256), dim3(256), 0, stream>>>(
        L[i].W, L[i].b, L[i].g, L[i].be, L[i].mn, L[i].vr,
        L[i].Oreal, Opad[i], L[i].K, Kpad[i], Wf[i], Sc[i], Sh[i], (i != 11) ? 1 : 0);
  }

  transpose_pts_kernel<<<dim3((B * N * 3 + 255) / 256), dim3(256), 0, stream>>>(pts, xyz, B, N);

  // SA1: 2048 -> 128 centroids, r=0.2, nsample=64, MLP [3,64,64,128]
  fps_kernel<<<dim3(B), dim3(256), 0, stream>>>(xyz, N, 128, fps1);
  gather_xyz_kernel<<<dim3((B * 128 * 3 + 255) / 256), dim3(256), 0, stream>>>(xyz, fps1, l1xyz, B, N, 128);
  sa_fused_kernel<<<dim3(B * 128), dim3(128), 0, stream>>>(
      xyz, nullptr, 0, l1xyz, B, N, 128, 64, 0.04f, 3, 32,
      Wf[0], Sc[0], Sh[0], 64,
      Wf[1], Sc[1], Sh[1], 64,
      Wf[2], Sc[2], Sh[2], 128,
      l1pts, 0);

  // SA2: 128 -> 32 centroids, r=0.4, nsample=64, MLP [131,128,128,256]
  fps_kernel<<<dim3(B), dim3(256), 0, stream>>>(l1xyz, 128, 32, fps2);
  gather_xyz_kernel<<<dim3((B * 32 * 3 + 255) / 256), dim3(256), 0, stream>>>(l1xyz, fps2, l2xyz, B, 128, 32);
  sa_fused_kernel<<<dim3(B * 32), dim3(128), 0, stream>>>(
      l1xyz, l1pts, 128, l2xyz, B, 128, 32, 64, 0.16f, 131, 160,
      Wf[3], Sc[3], Sh[3], 128,
      Wf[4], Sc[4], Sh[4], 128,
      Wf[5], Sc[5], Sh[5], 256,
      l2pts, 0);

  // SA3: group-all over 32 points, MLP [259,256,256,512]
  sa_fused_kernel<<<dim3(B), dim3(128), 0, stream>>>(
      l2xyz, l2pts, 256, nullptr, B, 32, 1, 32, 0.f, 259, 288,
      Wf[6], Sc[6], Sh[6], 256,
      Wf[7], Sc[7], Sh[7], 256,
      Wf[8], Sc[8], Sh[8], 512,
      gfeat, 1);

  // FC head
  build_fc_in_kernel<<<dim3((B * 544 + 255) / 256), dim3(256), 0, stream>>>(gfeat, one_hot, fcin, B);
  fc_gemm_kernel<<<dim3(64), dim3(128), 0, stream>>>(fcin, B, 544, Wf[9],  Sc[9],  Sh[9],  512, 512, nullptr, fc1o, 1);
  fc_gemm_kernel<<<dim3(32), dim3(128), 0, stream>>>(fc1o, B, 512, Wf[10], Sc[10], Sh[10], 256, 256, nullptr, fc2o, 1);
  fc_gemm_kernel<<<dim3(8),  dim3(128), 0, stream>>>(fc2o, B, 256, Wf[11], Sc[11], Sh[11], 64,  59,  (float*)d_out, nullptr, 0);
}
